// EMAVectorQuantizer_29609504539292
// MI455X (gfx1250) — compile-verified, matrix operand order checked by reading, not run on hardware
//
#include <hip/hip_runtime.h>
#include <hip/hip_bf16.h>

// ---------------------------------------------------------------------------
// VQ codebook lookup for MI455X (gfx1250, wave32, WMMA, async-to-LDS).
//
//   z:         [32, 256, 32, 32] f32   -> z_flat [N=32768, d=256]
//   embedding: [8192, 256] f32
//   out:       z_q [32,256,32,32] f32  ++  argmin indices [32768]
//
// argmin_k ( ||z||^2 + ||e_k||^2 - 2 z.e_k ) == argmin_k ( ||e_k||^2 - 2 z.e_k )
// Fused f16-in/f32-acc WMMA GEMM + argmin; embedding tiles streamed to LDS
// with global_load_async_to_lds_b128 (ASYNCcnt), double buffered, K split in
// two across blockIdx.y for parallelism, combined in a tiny final pass.
// ---------------------------------------------------------------------------

typedef __attribute__((ext_vector_type(16))) _Float16 v16h;
typedef __attribute__((ext_vector_type(8)))  _Float16 v8h;
typedef __attribute__((ext_vector_type(8)))  float    v8f;

union V16U { v16h v; v8h h[2]; };

#define NPTS    32768            // B*H*W
#define DIM     256
#define KEMB    8192
#define NTILES  (KEMB / 16)      // 512
#define KSPLIT  2
#define TILES_PER_SPLIT (NTILES / KSPLIT)   // 256

// ---------------------------------------------------------------------------
// Kernel 1: embedding f32 -> f16 (row major) + e2[k] = ||e_k||^2
// ---------------------------------------------------------------------------
__global__ void vq_prep_emb(const float* __restrict__ emb,
                            _Float16* __restrict__ ef16,
                            float* __restrict__ e2) {
  const int wave = threadIdx.x >> 5;
  const int lane = threadIdx.x & 31;
  const int row  = blockIdx.x * 8 + wave;

  const float* src = emb + (size_t)row * DIM + lane * 8;
  float4 f0 = *(const float4*)(src);
  float4 f1 = *(const float4*)(src + 4);

  float s = f0.x*f0.x + f0.y*f0.y + f0.z*f0.z + f0.w*f0.w
          + f1.x*f1.x + f1.y*f1.y + f1.z*f1.z + f1.w*f1.w;

  v8h h;
  h[0] = (_Float16)f0.x; h[1] = (_Float16)f0.y;
  h[2] = (_Float16)f0.z; h[3] = (_Float16)f0.w;
  h[4] = (_Float16)f1.x; h[5] = (_Float16)f1.y;
  h[6] = (_Float16)f1.z; h[7] = (_Float16)f1.w;
  *(v8h*)(ef16 + (size_t)row * DIM + lane * 8) = h;

  #pragma unroll
  for (int m = 16; m >= 1; m >>= 1) s += __shfl_xor(s, m, 32);
  if (lane == 0) e2[row] = s;
}

// ---------------------------------------------------------------------------
// Kernel 2: z [32][256][1024] f32 -> z_f16 [n][c] f16 (LDS-tiled transpose).
// ---------------------------------------------------------------------------
__global__ void vq_prep_z(const float* __restrict__ z,
                          _Float16* __restrict__ zf16) {
  __shared__ float sh[32][33];
  const int b  = blockIdx.z;
  const int ct = blockIdx.y;
  const int ht = blockIdx.x;
  const int tx = threadIdx.x & 31;
  const int ty = threadIdx.x >> 5;

  #pragma unroll
  for (int i = 0; i < 4; ++i) {
    int c = ct * 32 + ty + i * 8;
    sh[ty + i * 8][tx] = z[((size_t)(b * 256 + c)) * 1024 + ht * 32 + tx];
  }
  __syncthreads();
  #pragma unroll
  for (int i = 0; i < 4; ++i) {
    int hw = ht * 32 + ty + i * 8;
    size_t n = (size_t)b * 1024 + hw;
    zf16[n * DIM + ct * 32 + tx] = (_Float16)sh[tx][ty + i * 8];
  }
}

// ---------------------------------------------------------------------------
// Kernel 3: fused GEMM + partial argmin over one K-half.
// grid (128 m-blocks, 2 k-halves) x 256 threads (8 waves).
// Each wave owns TWO 16-row M tiles (A in 128 VGPRs) -> 2 WMMAs per B-frag
// ds_read. Embedding tiles (16x256 f16 = 8 KB) stream into double-buffered
// LDS via global_load_async_to_lds_b128; compute overlaps the async copy;
// one s_barrier + one s_wait_asynccnt per tile.
// ---------------------------------------------------------------------------
__device__ inline void async_copy_tile(const _Float16* gsrc, void* lds_dst) {
  // 8 KB tile, 256 threads x 2 x b128; inst offset applies to BOTH the
  // global and LDS sides, so +4096 covers the second half of the tile.
  uint64_t g = (uint64_t)(uintptr_t)gsrc + (uint32_t)threadIdx.x * 16u;
  uint32_t l = (uint32_t)(uintptr_t)lds_dst + (uint32_t)threadIdx.x * 16u;
  asm volatile("global_load_async_to_lds_b128 %0, %1, off"
               :: "v"(l), "v"(g) : "memory");
  asm volatile("global_load_async_to_lds_b128 %0, %1, off offset:4096"
               :: "v"(l), "v"(g) : "memory");
}

__device__ inline v16h load_frag_g(const _Float16* base, int koff) {
  V16U u;
  u.h[0] = *(const v8h*)(base + koff);
  u.h[1] = *(const v8h*)(base + 16 + koff);
  return u.v;
}

__global__ void __launch_bounds__(256, 1)
vq_gemm_argmin(const _Float16* __restrict__ zf16,
               const _Float16* __restrict__ ef16,
               const float* __restrict__ e2,
               float* __restrict__ pv,      // [KSPLIT][NPTS] partial min value
               int* __restrict__ pi) {      // [KSPLIT][NPTS] partial min index
  __shared__ __align__(16) _Float16 Btile[2][16 * DIM];   // 2 x 8 KB
  __shared__ __align__(16) float e2s[TILES_PER_SPLIT * 16];  // 16 KB

  const int lane     = threadIdx.x & 31;
  const int wave     = threadIdx.x >> 5;
  const int half_sel = lane >> 4;
  const int lid      = lane & 15;
  const int koff     = half_sel * 8;
  const int khalf    = blockIdx.y;
  const int tbase    = khalf * TILES_PER_SPLIT;

  // Kick off the async copy of the first tile immediately.
  async_copy_tile(ef16 + (size_t)tbase * 16 * DIM, &Btile[0][0]);

  // Stage this K-half's e2 (4096 floats) into LDS.
  {
    const float4* s = (const float4*)(e2 + tbase * 16);
    float4* d = (float4*)e2s;
    #pragma unroll
    for (int i = 0; i < 4; ++i) d[threadIdx.x + i * 256] = s[threadIdx.x + i * 256];
  }

  // Two resident A tiles per wave (rows m0.., m0+16..).
  const int m0 = blockIdx.x * 256 + wave * 32;
  const _Float16* arow0 = zf16 + (size_t)(m0 + lid) * DIM;
  const _Float16* arow1 = zf16 + (size_t)(m0 + 16 + lid) * DIM;
  v16h a0[8], a1[8];
  #pragma unroll
  for (int kc = 0; kc < 8; ++kc) {
    a0[kc] = load_frag_g(arow0 + kc * 32, koff);
    a1[kc] = load_frag_g(arow1 + kc * 32, koff);
  }

  float minv0[8], minv1[8];
  int   mini0[8], mini1[8];
  #pragma unroll
  for (int j = 0; j < 8; ++j) {
    minv0[j] = 3.0e38f; mini0[j] = 0;
    minv1[j] = 3.0e38f; mini1[j] = 0;
  }

  for (int tt = 0; tt < TILES_PER_SPLIT; ++tt) {
    const int buf = tt & 1;
    // Wait for my async loads of tile tt, then sync so everyone's landed
    // (and everyone is done reading buf^1 from iteration tt-1).
    asm volatile("s_wait_asynccnt 0x0" ::: "memory");
    __syncthreads();
    if (tt + 1 < TILES_PER_SPLIT)   // overlap next copy with this compute
      async_copy_tile(ef16 + (size_t)(tbase + tt + 1) * 16 * DIM, &Btile[buf ^ 1][0]);

    v8f c0 = {}, c1 = {};
    const _Float16* brow = &Btile[buf][0] + lid * DIM;
    #pragma unroll
    for (int kc = 0; kc < 8; ++kc) {
      V16U u;
      u.h[0] = *(const v8h*)(brow + kc * 32 + koff);        // ds_read_b128
      u.h[1] = *(const v8h*)(brow + kc * 32 + 16 + koff);   // ds_read_b128
      c0 = __builtin_amdgcn_wmma_f32_16x16x32_f16(
             false, a0[kc], false, u.v, (short)0, c0, false, false);
      c1 = __builtin_amdgcn_wmma_f32_16x16x32_f16(
             false, a1[kc], false, u.v, (short)0, c1, false, false);
    }

    const int   kidx = (tbase + tt) * 16 + lid;
    const float e2v  = e2s[tt * 16 + lid];
    #pragma unroll
    for (int j = 0; j < 8; ++j) {
      float s0 = fmaf(-2.0f, c0[j], e2v);
      float s1 = fmaf(-2.0f, c1[j], e2v);
      if (s0 < minv0[j]) { minv0[j] = s0; mini0[j] = kidx; }
      if (s1 < minv1[j]) { minv1[j] = s1; mini1[j] = kidx; }
    }
  }

  // Cross-lane argmin within each 16-lane group, lowest-index tie-break.
  #pragma unroll
  for (int m = 1; m < 16; m <<= 1) {
    #pragma unroll
    for (int j = 0; j < 8; ++j) {
      float ov; int oi;
      ov = __shfl_xor(minv0[j], m, 32); oi = __shfl_xor(mini0[j], m, 32);
      if (ov < minv0[j] || (ov == minv0[j] && oi < mini0[j])) { minv0[j] = ov; mini0[j] = oi; }
      ov = __shfl_xor(minv1[j], m, 32); oi = __shfl_xor(mini1[j], m, 32);
      if (ov < minv1[j] || (ov == minv1[j] && oi < mini1[j])) { minv1[j] = ov; mini1[j] = oi; }
    }
  }

  if (lid == 0) {
    const int rbase = m0 + half_sel * 8;
    #pragma unroll
    for (int j = 0; j < 8; ++j) {
      pv[khalf * NPTS + rbase + j]      = minv0[j];
      pi[khalf * NPTS + rbase + j]      = mini0[j];
      pv[khalf * NPTS + rbase + 16 + j] = minv1[j];
      pi[khalf * NPTS + rbase + 16 + j] = mini1[j];
    }
  }
}

// ---------------------------------------------------------------------------
// Kernel 3b: combine the two K-half candidates per row.
// ---------------------------------------------------------------------------
__global__ void vq_combine(const float* __restrict__ pv,
                           const int* __restrict__ pi,
                           int* __restrict__ idx_ws,
                           float* __restrict__ out_idx_f) {
  int n = blockIdx.x * 256 + threadIdx.x;
  float v0 = pv[n], v1 = pv[NPTS + n];
  int   i0 = pi[n], i1 = pi[NPTS + n];
  int best = (v1 < v0 || (v1 == v0 && i1 < i0)) ? i1 : i0;
  idx_ws[n]    = best;
  out_idx_f[n] = (float)best;
}

// ---------------------------------------------------------------------------
// Kernel 4: z_q[b][c][h][w] = embedding[idx[n]][c]  (f32 source).
// ---------------------------------------------------------------------------
__global__ void vq_gather(const float* __restrict__ emb,
                          const int* __restrict__ idx,
                          float* __restrict__ out) {
  size_t o  = (size_t)blockIdx.x * 256 + threadIdx.x;
  int hw    = (int)(o & 1023);
  size_t tt = o >> 10;
  int cc    = (int)(tt & 255);
  int b     = (int)(tt >> 8);
  int n     = b * 1024 + hw;
  out[o]    = emb[(size_t)idx[n] * DIM + cc];
}

// ---------------------------------------------------------------------------
extern "C" void kernel_launch(void* const* d_in, const int* in_sizes, int n_in,
                              void* d_out, int out_size, void* d_ws, size_t ws_size,
                              hipStream_t stream) {
  const float* z   = (const float*)d_in[0];   // [32,256,32,32]
  const float* emb = (const float*)d_in[1];   // [8192,256]
  float* out = (float*)d_out;                 // z_q (8388608 f32) ++ indices (32768)

  char* ws = (char*)d_ws;
  _Float16* zf16 = (_Float16*)(ws);                                   // 16 MB
  _Float16* ef16 = (_Float16*)(ws + ((size_t)16 << 20));              //  4 MB
  float*    e2   = (float*)   (ws + ((size_t)20 << 20));              // 32 KB
  float*    pv   = (float*)   (ws + ((size_t)20 << 20) + (1 << 15));  // 256 KB
  int*      pi   = (int*)     (ws + ((size_t)20 << 20) + (1 << 15) + (KSPLIT * NPTS * 4));
  int*      idxb = (int*)     (ws + ((size_t)20 << 20) + (1 << 15) + (2 * KSPLIT * NPTS * 4));

  // 1) embedding -> f16 + ||e||^2
  vq_prep_emb<<<KEMB / 8, 256, 0, stream>>>(emb, ef16, e2);

  // 2) z -> f16 [n][d] (transpose)
  vq_prep_z<<<dim3(32, 8, 32), 256, 0, stream>>>(z, zf16);

  // 3) WMMA GEMM + fused partial argmin (K split in 2)
  vq_gemm_argmin<<<dim3(NPTS / 256, KSPLIT), 256, 0, stream>>>(zf16, ef16, e2, pv, pi);

  // 3b) combine K-half candidates
  vq_combine<<<NPTS / 256, 256, 0, stream>>>(pv, pi, idxb,
                                             out + (size_t)32 * 256 * 32 * 32);

  // 4) gather codebook rows into NCHW z_q
  vq_gather<<<(32 * 256 * 32 * 32) / 256, 256, 0, stream>>>(emb, idxb, out);
}
